// ExpertGroup_10222022165064
// MI455X (gfx1250) — matrix operand order repaired and checked by reading, not compiled, not run on hardware
//
#include <hip/hip_runtime.h>

// ---------------------------------------------------------------------------
// Grouped MoE expert MLP for MI455X (gfx1250, wave32, WMMA, async->LDS).
//   out[t,:] = relu(x[t,:] @ w_up[e].T)^2 @ w_down[e].T   (bf16 compute)
// Passes: (0) f32->bf16 into ws, (1) WMMA GEMM1 + relu^2 -> bf16 H in ws,
// (2) WMMA GEMM2 -> f32 out. GEMM tiles double-buffered in LDS (108 KB of
// the 320 KB/WGP), filled with GLOBAL_LOAD_ASYNC_TO_LDS_B128 (ASYNCcnt) so
// the copy needs no VGPR staging and overlaps the 32 WMMAs per K tile.
// ---------------------------------------------------------------------------

typedef __bf16 bf16_t;
typedef __attribute__((ext_vector_type(8)))  __bf16 v8bf;
typedef __attribute__((ext_vector_type(16))) __bf16 v16bf;
typedef __attribute__((ext_vector_type(8)))  float  v8f;

#define T_TOK 8192
#define D_DIM 1024
#define H_DIM 2048
#define N_EXP 8

#define BM 128              // block rows (tokens)
#define BN 256              // block cols
#define KS 64               // K tile staged in LDS per step
#define LDS_P 72            // padded LDS row stride (elems): 64 + 8 -> ds_load
                            // start banks 4*(9r mod 16): conflict-free per half
#define A_ELEMS (BM * LDS_P)            // 9216
#define B_ELEMS (BN * LDS_P)            // 18432
#define BUF_ELEMS (A_ELEMS + B_ELEMS)   // 27648
#define LDS_BYTES (2 * BUF_ELEMS * (int)sizeof(bf16_t))  // 110592 B

// Async global->LDS 16B copy (gfx1250): VDST = LDS byte address VGPR,
// VADDR = 64-bit global address. Tracked by ASYNCcnt. Generic pointers to
// LDS carry the byte offset in addr[31:0], so truncation yields the DS addr.
__device__ __forceinline__ void async_copy16(void* lds_dst, const void* gsrc) {
  unsigned l = (unsigned)(unsigned long long)lds_dst;
  asm volatile("global_load_async_to_lds_b128 %0, %1, off"
               :: "v"(l), "v"(gsrc) : "memory");
}
__device__ __forceinline__ void wait_async0() {
  asm volatile("s_wait_asynccnt 0x0" ::: "memory");
}

// -------- pass 0: f32 -> bf16 (RNE), 8 elems/thread -------------------------
__global__ __launch_bounds__(256) void cvt_f32_to_bf16(
    const float* __restrict__ in, bf16_t* __restrict__ out, size_t n) {
  size_t i = ((size_t)blockIdx.x * blockDim.x + threadIdx.x) * 8;
  if (i + 8 <= n) {
    const float4* p = reinterpret_cast<const float4*>(in + i);
    float4 a = p[0];
    float4 b = p[1];
    v8bf o;
    o[0] = (bf16_t)a.x; o[1] = (bf16_t)a.y; o[2] = (bf16_t)a.z; o[3] = (bf16_t)a.w;
    o[4] = (bf16_t)b.x; o[5] = (bf16_t)b.y; o[6] = (bf16_t)b.z; o[7] = (bf16_t)b.w;
    *reinterpret_cast<v8bf*>(out + i) = o;
  }
}

// -------- LDS fragment loads (CDNA5 WMMA bf16 VGPR layouts, wave32) ---------
// A (16x32, MxK): lane half h: K=[8h..8h+7] in VGPR0-3, K=[16+8h..] in VGPR4-7.
__device__ __forceinline__ v16bf lds_frag_a(const bf16_t* buf, int row_local,
                                            int kk, int half) {
  const bf16_t* p = buf + row_local * LDS_P + kk + half * 8;
  v8bf lo = *reinterpret_cast<const v8bf*>(p);
  v8bf hi = *reinterpret_cast<const v8bf*>(p + 16);
  return __builtin_shufflevector(lo, hi, 0, 1, 2, 3, 4, 5, 6, 7,
                                 8, 9, 10, 11, 12, 13, 14, 15);
}
// B (32x16, KxN): lane half h: K=[8h..8h+7] in VGPR0-3, K=[16+8h..] in VGPR4-7.
__device__ __forceinline__ v16bf lds_frag_b(const bf16_t* buf, int row_local,
                                            int kk, int half) {
  const bf16_t* p = buf + row_local * LDS_P + kk + half * 8;
  v8bf lo = *reinterpret_cast<const v8bf*>(p);
  v8bf hi = *reinterpret_cast<const v8bf*>(p + 16);
  return __builtin_shufflevector(lo, hi, 0, 1, 2, 3, 4, 5, 6, 7,
                                 8, 9, 10, 11, 12, 13, 14, 15);
}

// -------- WMMA GEMM over one expert segment ---------------------------------
// C[M,N] = A[M,K] @ B[e][N,K]^T. 256 threads = 8 waves (2 m x 4 n), block
// tile 128x256, wave tile 64x64 = 4x4 WMMA accumulators. K staged in LDS in
// steps of 64, double buffered via async copies; one barrier per K tile.
template <int KDIM, int NDIM, bool SQREL>
__global__ __launch_bounds__(256) void moe_gemm_wmma(
    const bf16_t* __restrict__ A,        // (T_TOK, KDIM)
    const bf16_t* __restrict__ B,        // (N_EXP, NDIM, KDIM)
    const int*    __restrict__ counts,   // (N_EXP,)
    bf16_t*       __restrict__ out_bf,   // (T_TOK, NDIM) if SQREL
    float*        __restrict__ out_f)    // (T_TOK, NDIM) if !SQREL
{
  extern __shared__ bf16_t smem[];       // [2][A_ELEMS + B_ELEMS]

  const int e = blockIdx.z;
  int start = 0;
  for (int i = 0; i < e; ++i) start += counts[i];   // scalar prefix sum (E=8)
  const int cnt = counts[e];
  const int m_blk = blockIdx.y * BM;
  if (m_blk >= cnt) return;                         // uniform early-out
  const int seg_end = start + cnt;

  const int tid  = threadIdx.x;
  const int lane = tid & 31;
  const int wave = tid >> 5;
  const int wm = wave & 1;          // 0..1
  const int wn = wave >> 1;         // 0..3
  const int half = lane >> 4;       // 0..1
  const int r = lane & 15;          // 0..15

  const int row0 = start + m_blk;                 // block's first token row
  const int n0   = blockIdx.x * BN;               // block's first column
  const bf16_t* __restrict__ Be = B + (size_t)e * NDIM * KDIM;

  v8f acc[4][4];
#pragma unroll
  for (int i = 0; i < 4; ++i)
#pragma unroll
    for (int j = 0; j < 4; ++j)
      acc[i][j] = (v8f){0.f, 0.f, 0.f, 0.f, 0.f, 0.f, 0.f, 0.f};

  // Async staging: per thread 4 A-vectors + 8 B-vectors of 8 bf16 per K tile,
  // copied straight to LDS (no VGPR data path).
  auto async_tile = [&](int k0, bf16_t* buf) {
    bf16_t* bufA = buf;
    bf16_t* bufB = buf + A_ELEMS;
#pragma unroll
    for (int i = 0; i < 4; ++i) {
      const int vi = tid + 256 * i;               // 0..1023
      int grow = row0 + (vi >> 3);                // rows may run past segment
      grow = grow < T_TOK ? grow : T_TOK - 1;     // clamp; masked at store
      async_copy16(bufA + (vi >> 3) * LDS_P + (vi & 7) * 8,
                   A + (size_t)grow * KDIM + k0 + (vi & 7) * 8);
    }
#pragma unroll
    for (int i = 0; i < 8; ++i) {
      const int vi = tid + 256 * i;               // 0..2047
      async_copy16(bufB + (vi >> 3) * LDS_P + (vi & 7) * 8,
                   Be + (size_t)(n0 + (vi >> 3)) * KDIM + k0 + (vi & 7) * 8);
    }
  };

  constexpr int nK = KDIM / KS;

  async_tile(0, smem);
  wait_async0();
  __syncthreads();

  for (int kt = 0; kt < nK; ++kt) {
    const bf16_t* bufA = smem + (kt & 1) * BUF_ELEMS;
    const bf16_t* bufB = bufA + A_ELEMS;

    // Issue next tile's copies into the other buffer. Safe: last reads of
    // buf[(kt+1)&1] happened in iter kt-1 and were sealed by its barrier.
    if (kt + 1 < nK) async_tile((kt + 1) * KS, smem + ((kt + 1) & 1) * BUF_ELEMS);

#pragma unroll
    for (int kk = 0; kk < KS; kk += 32) {
      v16bf af[4], bf[4];
#pragma unroll
      for (int i = 0; i < 4; ++i)
        af[i] = lds_frag_a(bufA, wm * 64 + i * 16 + r, kk, half);
#pragma unroll
      for (int j = 0; j < 4; ++j)
        bf[j] = lds_frag_b(bufB, wn * 64 + j * 16 + r, kk, half);
#pragma unroll
      for (int i = 0; i < 4; ++i)
#pragma unroll
        for (int j = 0; j < 4; ++j)
          acc[i][j] = __builtin_amdgcn_wmma_f32_16x16x32_bf16(
              false, af[i], false, bf[j], (short)0, acc[i][j], false, false);
    }

    if (kt + 1 < nK) {
      wait_async0();      // own copies into buf[(kt+1)&1] complete
      __syncthreads();    // visible to all waves before next iteration reads
    }
  }

  // Epilogue. C/D layout: N = n0 + r, M = m0 + v + 8*half. Mask rows to the
  // expert segment: segments partition [0,T), each row written exactly once.
  const int m_wave = row0 + wm * 64;
  const int n_wave = n0 + wn * 64;
#pragma unroll
  for (int i = 0; i < 4; ++i) {
#pragma unroll
    for (int v = 0; v < 8; ++v) {
      const int row = m_wave + i * 16 + v + 8 * half;
      if (row < seg_end) {
#pragma unroll
        for (int j = 0; j < 4; ++j) {
          const int col = n_wave + j * 16 + r;
          float val = acc[i][j][v];
          if constexpr (SQREL) {
            val = val > 0.f ? val : 0.f;
            val = val * val;
            out_bf[(size_t)row * NDIM + col] = (bf16_t)val;
          } else {
            out_f[(size_t)row * NDIM + col] = val;
          }
        }
      }
    }
  }
}

// ---------------------------------------------------------------------------
extern "C" void kernel_launch(void* const* d_in, const int* in_sizes, int n_in,
                              void* d_out, int out_size, void* d_ws, size_t ws_size,
                              hipStream_t stream) {
  const float* x      = (const float*)d_in[0];  // (T, D) f32
  const int*   counts = (const int*)d_in[1];    // (E,)   i32
  const float* w_up   = (const float*)d_in[2];  // (E, H, D) f32
  const float* w_down = (const float*)d_in[3];  // (E, D, H) f32
  float*       out    = (float*)d_out;          // (T, D) f32

  const size_t n_x  = (size_t)T_TOK * D_DIM;          // 8M
  const size_t n_wu = (size_t)N_EXP * H_DIM * D_DIM;  // 16M
  const size_t n_wd = (size_t)N_EXP * D_DIM * H_DIM;  // 16M

  bf16_t* xb  = (bf16_t*)d_ws;       // T x D     bf16 (16 MB)
  bf16_t* wub = xb + n_x;            // E x H x D bf16 (32 MB)
  bf16_t* wdb = wub + n_wu;          // E x D x H bf16 (32 MB)
  bf16_t* hb  = wdb + n_wd;          // T x H     bf16 (32 MB)

  // Pass 0: f32 -> bf16 (all element counts are multiples of 8*256)
  cvt_f32_to_bf16<<<dim3((unsigned)(n_x / (8 * 256))), 256, 0, stream>>>(x, xb, n_x);
  cvt_f32_to_bf16<<<dim3((unsigned)(n_wu / (8 * 256))), 256, 0, stream>>>(w_up, wub, n_wu);
  cvt_f32_to_bf16<<<dim3((unsigned)(n_wd / (8 * 256))), 256, 0, stream>>>(w_down, wdb, n_wd);

  // Pass 1: H = relu(x @ w_up[e]^T)^2  (bf16), K=D, N=H
  {
    dim3 grid(H_DIM / BN, T_TOK / BM, N_EXP);  // (8, 64, 8)
    moe_gemm_wmma<D_DIM, H_DIM, true>
        <<<grid, 256, LDS_BYTES, stream>>>(xb, wub, counts, hb, nullptr);
  }
  // Pass 2: out = H @ w_down[e]^T  (f32), K=H, N=D
  {
    dim3 grid(D_DIM / BN, T_TOK / BM, N_EXP);  // (4, 64, 8)
    moe_gemm_wmma<H_DIM, D_DIM, false>
        <<<grid, 256, LDS_BYTES, stream>>>(hb, wdb, counts, nullptr, out);
  }
}
